// ValueModel_83708912599091
// MI455X (gfx1250) — matrix operand order repaired and checked by reading, not compiled, not run on hardware
//
#include <hip/hip_runtime.h>

typedef __attribute__((ext_vector_type(16))) _Float16 v16h;
typedef __attribute__((ext_vector_type(8)))  float    v8f;
typedef __attribute__((ext_vector_type(4)))  float    f32x4;
typedef __attribute__((ext_vector_type(4)))  int      i32x4;
// 4-byte-aligned view for b128 loads from odd-stride (59-float) rows;
// gfx1250 DWORD alignment mode allows dword-aligned b128.
typedef f32x4 __attribute__((aligned(4))) f32x4u;

#define N_IN        59
#define N_VALUES    12
#define THREADS     256
#define ROWS_PER_BLOCK 128          // 8 waves * 16 rows
#define KPAD        64
#define LDS_A_STRIDE 68             // floats; 272B rows -> 16B aligned, bank = (4*row+k)%64 conflict-free
#define LDS_F_STRIDE 13             // floats; conflict-free feature staging

__global__ __launch_bounds__(THREADS)
void value_model_kernel(const float* __restrict__ na,
                        const int*   __restrict__ cards,     // [B,4,3] int32
                        const float* __restrict__ W,         // [12,59]
                        const float* __restrict__ bias,      // [12]
                        float*       __restrict__ out,       // [B,4]
                        int B)
{
    __shared__ float ldsA[ROWS_PER_BLOCK * LDS_A_STRIDE];    // 34816 B
    __shared__ float ldsF[ROWS_PER_BLOCK * LDS_F_STRIDE];    //  6656 B

    const int tid       = threadIdx.x;
    const int lane      = tid & 31;
    const int wave      = tid >> 5;
    const int blockRow0 = blockIdx.x * ROWS_PER_BLOCK;
    const bool fullTile = (blockRow0 + ROWS_PER_BLOCK) <= B; // block-uniform

    // Prefetch this block's cards slice (gfx1250 global_prefetch path);
    // 128 rows * 12 idx * 4B = 6144 B, one address per thread 24B apart.
    __builtin_prefetch((const char*)(cards + (size_t)blockRow0 * 12) + tid * 24, 0, 1);

    // ---- Stage A tile into LDS, zero-padded K 59..63 (rows >= B zeroed) ----
    // Zero pad columns first.
    for (int r = tid; r < ROWS_PER_BLOCK; r += THREADS) {
        #pragma unroll
        for (int k = N_IN; k < KPAD; ++k)
            ldsA[r * LDS_A_STRIDE + k] = 0.0f;
    }
    if (fullTile) {
        // 14 x b128 quads per row cover k=0..55 (coalesced 16B/lane, streaming NT).
        for (int s = tid; s < ROWS_PER_BLOCK * 14; s += THREADS) {
            int r = s / 14;
            int q = s - r * 14;
            const f32x4u* src =
                (const f32x4u*)(na + (size_t)(blockRow0 + r) * N_IN + q * 4);
            f32x4 v = __builtin_nontemporal_load(src);
            *(f32x4*)&ldsA[r * LDS_A_STRIDE + q * 4] = v;   // 16B-aligned LDS store
        }
        // Tail elements k=56..58.
        for (int i = tid; i < ROWS_PER_BLOCK * 3; i += THREADS) {
            int r = i / 3;
            int k = 56 + (i - r * 3);
            ldsA[r * LDS_A_STRIDE + k] =
                __builtin_nontemporal_load(na + (size_t)(blockRow0 + r) * N_IN + k);
        }
    } else {
        // Partial last block: safe scalar path with row bounds check.
        for (int i = tid; i < ROWS_PER_BLOCK * N_IN; i += THREADS) {
            int r = i / N_IN;
            int k = i - r * N_IN;
            int gr = blockRow0 + r;
            ldsA[r * LDS_A_STRIDE + k] = (gr < B) ? na[(size_t)gr * N_IN + k] : 0.0f;
        }
    }
    __syncthreads();

    // ---- Build B fragments from W (tiny, cache-resident) ----
    // B layout (32x16, KxN), v16h per lane: half h -> K = h + 8*(h>=8) + 8*(lane>=16), N = lane&15
    const int n     = lane & 15;
    const int kbase = (lane >> 4) * 8;
    v16h bfrag0, bfrag1;
    #pragma unroll
    for (int h = 0; h < 16; ++h) {
        int k0 = kbase + h + ((h >= 8) ? 8 : 0);           // step 0: K = k0
        int k1 = 32 + kbase + h + ((h >= 8) ? 8 : 0);      // step 1: K = k1
        float w0 = (k0 < N_IN && n < N_VALUES) ? W[n * N_IN + k0] : 0.0f;
        float w1 = (k1 < N_IN && n < N_VALUES) ? W[n * N_IN + k1] : 0.0f;
        bfrag0[h] = (_Float16)w0;
        bfrag1[h] = (_Float16)w1;
    }

    // ---- A fragments from LDS + 2x WMMA (K = 0..31, 32..63) ----
    // A layout (16x32, MxK): M = lane&15; half h -> K = h + 8*(h>=8) + 8*(lane>=16)
    const int m       = lane & 15;
    const int rowBase = wave * 16;
    const f32x4* arow4 = (const f32x4*)&ldsA[(rowBase + m) * LDS_A_STRIDE];
    const int kla = (lane < 16) ? 0 : 8;

    v8f acc = {};
    #pragma unroll
    for (int s = 0; s < 2; ++s) {
        f32x4 q[4];
        q[0] = arow4[(32 * s + kla) >> 2];
        q[1] = arow4[((32 * s + kla) >> 2) + 1];
        q[2] = arow4[(32 * s + 16 + kla) >> 2];
        q[3] = arow4[((32 * s + 16 + kla) >> 2) + 1];
        const float* qa = (const float*)q;
        v16h afrag;
        #pragma unroll
        for (int h = 0; h < 16; ++h)
            afrag[h] = (_Float16)qa[h];
        if (s == 0)
            acc = __builtin_amdgcn_wmma_f32_16x16x32_f16(false, afrag, false, bfrag0,
                                                         (short)0, acc, false, false);
        else
            acc = __builtin_amdgcn_wmma_f32_16x16x32_f16(false, afrag, false, bfrag1,
                                                         (short)0, acc, false, false);
    }

    // ---- Add bias and stage features to LDS ----
    // C/D layout: acc[j] -> M = rowBase + j + 8*(lane>=16), N = lane&15
    if (n < N_VALUES) {
        float bn = bias[n];
        const int mhi = (lane < 16) ? 0 : 8;
        #pragma unroll
        for (int j = 0; j < 8; ++j) {
            int mrow = rowBase + j + mhi;
            ldsF[mrow * LDS_F_STRIDE + n] = acc[j] + bn;
        }
    }
    __syncthreads();

    // ---- Gather + per-card sum: one thread per row ----
    // 12 indices per row as 3x b128 (48B-aligned, NT), 4 outputs as one b128 NT store.
    if (tid < ROWS_PER_BLOCK) {
        int gr = blockRow0 + tid;
        if (gr < B) {
            const i32x4* c4 = (const i32x4*)(cards + (size_t)gr * 12);
            i32x4 q0 = __builtin_nontemporal_load(c4 + 0);
            i32x4 q1 = __builtin_nontemporal_load(c4 + 1);
            i32x4 q2 = __builtin_nontemporal_load(c4 + 2);
            const float* f = &ldsF[tid * LDS_F_STRIDE];
            f32x4 res;
            res.x = f[q0.x] + f[q0.y] + f[q0.z];
            res.y = f[q0.w] + f[q1.x] + f[q1.y];
            res.z = f[q1.z] + f[q1.w] + f[q2.x];
            res.w = f[q2.y] + f[q2.z] + f[q2.w];
            __builtin_nontemporal_store(res, (f32x4*)(out + (size_t)gr * 4));
        }
    }
}

extern "C" void kernel_launch(void* const* d_in, const int* in_sizes, int n_in,
                              void* d_out, int out_size, void* d_ws, size_t ws_size,
                              hipStream_t stream) {
    const float* na    = (const float*)d_in[0];   // [B,59] f32
    const int*   cards = (const int*)d_in[1];     // [B,4,3] int32
    const float* W     = (const float*)d_in[2];   // [12,59] f32
    const float* bias  = (const float*)d_in[3];   // [12] f32
    float*       out   = (float*)d_out;           // [B,4] f32

    int B = in_sizes[0] / N_IN;
    int grid = (B + ROWS_PER_BLOCK - 1) / ROWS_PER_BLOCK;
    hipLaunchKernelGGL(value_model_kernel, dim3(grid), dim3(THREADS), 0, stream,
                       na, cards, W, bias, out, B);
}
